// CrossAttention_42898133352567
// MI455X (gfx1250) — matrix operand order repaired and checked by reading, not compiled, not run on hardware
//
#include <hip/hip_runtime.h>
#include <cstdint>
#include <cstddef>

// ---------------------------------------------------------------------------
// CDNA5 / gfx1250 cross-attention pipeline
// wave32, WMMA f32_16x16x32_f16, async-to-LDS fills, TDM tile load
// ---------------------------------------------------------------------------

typedef _Float16 half_t;
typedef __attribute__((ext_vector_type(16))) _Float16 v16h;
typedef __attribute__((ext_vector_type(8)))  _Float16 v8h;
typedef __attribute__((ext_vector_type(8)))  float    v8f;
typedef __attribute__((ext_vector_type(4)))  unsigned int u32x4;
typedef __attribute__((ext_vector_type(8)))  unsigned int u32x8;

#define HEADS_  16
#define DHEAD_  64
#define TEXTL_  77
#define IMGL_   256
#define NQ_     1024
#define NB_     32
#define DMODEL_ 1024

__device__ __forceinline__ v8f zero8f() {
    v8f z; for (int i = 0; i < 8; ++i) z[i] = 0.f; return z;
}

// Async DMA: 16B global -> LDS, tracked by ASYNCcnt (no VGPR round trip).
__device__ __forceinline__ void async_load_b128(uint32_t lds_addr, const void* gptr) {
    asm volatile("global_load_async_to_lds_b128 %0, %1, off"
                 :: "v"(lds_addr), "v"(gptr) : "memory");
}
__device__ __forceinline__ void wait_async() {
    asm volatile("s_wait_asynccnt 0x0" ::: "memory");
}

// A-fragment (16 rows x K=32), per ISA 7.12.2 16-bit A layout:
// lane L: row = L&15; halves 0..7 = K (L>>4)*8+0..7 ; halves 8..15 = +16
__device__ __forceinline__ v16h frag_a(const half_t* p, int ld, int lane) {
    const int r = lane & 15, hh = lane >> 4;
    const half_t* p0 = p + r * ld + hh * 8;
    v8h lo = *(const v8h*)p0;
    v8h hi = *(const v8h*)(p0 + 16);
    return __builtin_shufflevector(lo, hi, 0,1,2,3,4,5,6,7,8,9,10,11,12,13,14,15);
}

// B-fragment (K=32 x N=16), N-major storage (row n holds its K values).
// lane L: col n = L&15; halves 0..7 = K (L>>4)*8+0..7 ; halves 8..15 = +16
__device__ __forceinline__ v16h frag_b(const half_t* p, int ld, int lane) {
    const int n = lane & 15, hh = lane >> 4;
    const half_t* p0 = p + n * ld + hh * 8;
    v8h lo = *(const v8h*)p0;
    v8h hi = *(const v8h*)(p0 + 16);
    return __builtin_shufflevector(lo, hi, 0,1,2,3,4,5,6,7,8,9,10,11,12,13,14,15);
}

__device__ __forceinline__ v8f wmma32(v16h a, v16h b, v8f c) {
    return __builtin_amdgcn_wmma_f32_16x16x32_f16(false, a, false, b, (short)0, c, false, false);
}

// ---------------------------------------------------------------------------
// Weight convert + transpose: dst[n][k] = (f16) src[k][n]   (1024x1024)
// ---------------------------------------------------------------------------
__global__ __launch_bounds__(256)
void ca_convert_w(const float* __restrict__ src, half_t* __restrict__ dst) {
    const int idx = blockIdx.x * 256 + threadIdx.x;   // 1M elements, grid=4096
    const int n = idx >> 10, k = idx & 1023;
    dst[idx] = (half_t)src[(size_t)k * DMODEL_ + n];
}

// ---------------------------------------------------------------------------
// Projection GEMM: Y = A(f32, batched rows) x W^T(f16) -> f16 head-split
// ---------------------------------------------------------------------------
__global__ __launch_bounds__(256)
void ca_proj_gemm(const float* __restrict__ A, const half_t* __restrict__ Wt,
                  half_t* __restrict__ out,
                  int rows_per_batch, int batch_stride, int row_off, int M_total) {
    __shared__ half_t As[128 * 40];   // stride 40 halves: conflict-free b128 frags
    __shared__ half_t Bs[128 * 40];

    const int tid = threadIdx.x, lane = tid & 31, wv = tid >> 5;
    const int m0 = blockIdx.x * 128, n0 = blockIdx.y * 128;
    const int wm = (wv & 3) * 32, wn = (wv >> 2) * 64;

    const int frow = tid >> 1, fc = (tid & 1) * 16;
    const int gr = m0 + frow;
    const bool valid = gr < M_total;
    int bb = 0, rr = 0;
    if (valid) { bb = gr / rows_per_batch; rr = gr - bb * rows_per_batch; }
    const float*  arow = A + ((size_t)bb * batch_stride + row_off + rr) * DMODEL_;
    const half_t* brow = Wt + (size_t)(n0 + frow) * DMODEL_;
    const uint32_t bs_lds = (uint32_t)(uintptr_t)(Bs + frow * 40 + fc);

    v8f acc[2][4];
    for (int mi = 0; mi < 2; ++mi)
        for (int ni = 0; ni < 4; ++ni) acc[mi][ni] = zero8f();

    for (int k0 = 0; k0 < DMODEL_; k0 += 32) {
        // ---- B tile: f16, straight copy -> async DMA to LDS ----
        async_load_b128(bs_lds,      brow + k0 + fc);
        async_load_b128(bs_lds + 16, brow + k0 + fc + 8);
        // ---- A tile: f32 -> f16 convert through VGPRs ----
        {
            half_t* d = As + frow * 40 + fc;
            if (valid) {
                const float4* s = (const float4*)(arow + k0 + fc);
                float4 f0 = s[0], f1 = s[1], f2 = s[2], f3 = s[3];
                v8h h0 = { (half_t)f0.x,(half_t)f0.y,(half_t)f0.z,(half_t)f0.w,
                           (half_t)f1.x,(half_t)f1.y,(half_t)f1.z,(half_t)f1.w };
                v8h h1 = { (half_t)f2.x,(half_t)f2.y,(half_t)f2.z,(half_t)f2.w,
                           (half_t)f3.x,(half_t)f3.y,(half_t)f3.z,(half_t)f3.w };
                *(v8h*)d = h0; *(v8h*)(d + 8) = h1;
            } else {
                v8h z = { (half_t)0,(half_t)0,(half_t)0,(half_t)0,
                          (half_t)0,(half_t)0,(half_t)0,(half_t)0 };
                *(v8h*)d = z; *(v8h*)(d + 8) = z;
            }
            if (valid && k0 + 256 < DMODEL_) __builtin_prefetch(arow + k0 + 256, 0, 0);
        }
        wait_async();
        __syncthreads();

        v16h a0 = frag_a(As + wm * 40,        40, lane);
        v16h a1 = frag_a(As + (wm + 16) * 40, 40, lane);
        for (int ni = 0; ni < 4; ++ni) {
            v16h bf = frag_b(Bs + (wn + ni * 16) * 40, 40, lane);
            acc[0][ni] = wmma32(a0, bf, acc[0][ni]);
            acc[1][ni] = wmma32(a1, bf, acc[1][ni]);
        }
        __syncthreads();
    }

    const int ccol = lane & 15, crow = (lane >> 4) * 8;
    for (int mi = 0; mi < 2; ++mi)
        for (int ni = 0; ni < 4; ++ni)
            for (int r = 0; r < 8; ++r) {
                const int row = m0 + wm + mi * 16 + crow + r;
                if (row < M_total) {
                    const int col = n0 + wn + ni * 16 + ccol;
                    const int b2 = row / rows_per_batch, r2 = row - b2 * rows_per_batch;
                    const int hh = col >> 6, dd = col & 63;
                    out[(((size_t)b2 * HEADS_ + hh) * rows_per_batch + r2) * DHEAD_ + dd] =
                        (half_t)acc[mi][ni][r];
                }
            }
}

// ---------------------------------------------------------------------------
// Fused attention per (128-query tile, head, batch); 8 waves x 16 queries.
// ---------------------------------------------------------------------------
__global__ __launch_bounds__(256)
void ca_attn(const half_t* __restrict__ qw,  const half_t* __restrict__ kw,
             const half_t* __restrict__ vw,  const half_t* __restrict__ kipw,
             const half_t* __restrict__ vipw,
             const float*  __restrict__ relk, const float* __restrict__ relv,
             const float*  __restrict__ alpha_p, half_t* __restrict__ attn_out) {
    extern __shared__ char smem_raw[];
    half_t* q_s   = (half_t*)smem_raw;          // [128][72]
    half_t* kt_s  = q_s   + 128 * 72;           // [80][72]   rows 77..79 zero
    half_t* vt_s  = kt_s  + 80 * 72;            // [64][88]   V^T (d-major)
    half_t* kip_s = vt_s  + 64 * 88;            // [256][72]  (filled by TDM)
    half_t* vip_s = kip_s + 256 * 72;           // [64][264]  V_ip^T
    half_t* rk_s  = vip_s + 64 * 264;           // [204][64]  rel_k window
    half_t* rv_s  = rk_s  + 204 * 64;           // [204][64]  rel_v window
    float*  sS    = (float*)(rv_s + 204 * 64);  // 8 x [16][96] score staging
    half_t* sP    = (half_t*)(sS + 8 * 16 * 96);// 8 x [16][104] prob staging
    float*  stats = (float*)(sP + 8 * 16 * 104);// 8 x 64 floats

    const int tid = threadIdx.x, lane = tid & 31, wv = tid >> 5;
    const int q0 = blockIdx.x * 128;
    const int h  = blockIdx.y, b = blockIdx.z;
    const float scale = 0.125f;                  // 64^-0.5
    const int   base  = 897 - q0;                // rel window base index

    // ---------------- LDS fills ----------------
    // K_ip tile (256x64 f16) via Tensor Data Mover: 2-D tile, LDS row padding
    // 4 DWORDs (16B) after every 32 DWORDs (128B) -> [256][72] layout.
    if (wv == 0) {
        const uint64_t ga = (uint64_t)(uintptr_t)
            (kipw + (((size_t)b * HEADS_ + h) * IMGL_) * DHEAD_);
        const uint32_t la = (uint32_t)(uintptr_t)kip_s;
        u32x4 g0;
        g0[0] = 1u;                                              // count=1, user D#
        g0[1] = la;                                              // lds_addr
        g0[2] = (uint32_t)ga;                                    // global_addr[31:0]
        g0[3] = (uint32_t)((ga >> 32) & 0x1FFFFFFu) | (2u << 30);// ga[56:32] | type=2
        u32x8 g1;
        g1[0] = (1u << 16) | (1u << 20) | (4u << 22) | (3u << 25);
        //        data_size=2B | pad_en | pad every 32 DW | pad 4 DW
        g1[1] = (uint32_t)DHEAD_ << 16;                          // tensor_dim0 @[79:48]
        g1[2] = (uint32_t)IMGL_  << 16;                          // tensor_dim1 @[111:80]
        g1[3] = (uint32_t)DHEAD_ << 16;                          // tile_dim0  @[127:112]
        g1[4] = (uint32_t)IMGL_;                                 // tile_dim1  @[143:128]
        g1[5] = (uint32_t)DHEAD_;                                // dim0_stride @[207:160]
        g1[6] = 0u;
        g1[7] = 0u;
        asm volatile("tensor_load_to_lds %0, %1, null, null"
                     :: "s"(g0), "s"(g1) : "memory");
    }

    { // Q tile (128x64 f16): async DMA, 1024 x 16B chunks
        const half_t* gq = qw + (((size_t)b * HEADS_ + h) * NQ_ + q0) * DHEAD_;
        const uint32_t qb = (uint32_t)(uintptr_t)q_s;
        for (int u = tid; u < 1024; u += 256) {
            const int row = u >> 3, c8 = (u & 7) * 8;
            async_load_b128(qb + (uint32_t)(row * 144 + c8 * 2),
                            gq + (size_t)row * DHEAD_ + c8);
        }
    }
    { // K text (77x64 f16): async DMA, 616 x 16B chunks
        const half_t* gk = kw + (((size_t)b * HEADS_ + h) * TEXTL_) * DHEAD_;
        const uint32_t kb = (uint32_t)(uintptr_t)kt_s;
        for (int u = tid; u < 616; u += 256) {
            const int row = u >> 3, c8 = (u & 7) * 8;
            async_load_b128(kb + (uint32_t)(row * 144 + c8 * 2),
                            gk + (size_t)row * DHEAD_ + c8);
        }
    }
    if (tid < 24) { // zero-pad K text rows 77..79 (cols 0..63)
        const int row = 77 + (tid >> 3), c8 = (tid & 7) * 8;
        v8h z = { (half_t)0,(half_t)0,(half_t)0,(half_t)0,
                  (half_t)0,(half_t)0,(half_t)0,(half_t)0 };
        *(v8h*)(kt_s + row * 72 + c8) = z;
    }
    if (tid < 160) { // V text, transposed into [d][j]
        const int j = tid >> 1, d0 = (tid & 1) * 32;
        half_t vv[32];
        if (j < TEXTL_) {
            const half_t* s = vw + (((size_t)b * HEADS_ + h) * TEXTL_ + j) * DHEAD_ + d0;
            for (int i = 0; i < 32; ++i) vv[i] = s[i];
        } else {
            for (int i = 0; i < 32; ++i) vv[i] = (half_t)0;
        }
        for (int i = 0; i < 32; ++i) vt_s[(d0 + i) * 88 + j] = vv[i];
    }
    for (int u = tid; u < 512; u += 256) { // V img transposed
        const int j = u >> 1, d0 = (u & 1) * 32;
        const half_t* s = vipw + (((size_t)b * HEADS_ + h) * IMGL_ + j) * DHEAD_ + d0;
        for (int i = 0; i < 32; ++i) vip_s[(d0 + i) * 264 + j] = s[i];
    }
    for (int u = tid; u < 408; u += 256) { // rel_k window (f32 -> f16)
        const int w = u >> 1, d0 = (u & 1) * 32;
        const float* s = relk + (size_t)(base + w) * DHEAD_ + d0;
        for (int i = 0; i < 32; ++i) rk_s[w * 64 + d0 + i] = (half_t)s[i];
    }
    for (int u = tid; u < 408; u += 256) { // rel_v window
        const int w = u >> 1, d0 = (u & 1) * 32;
        const float* s = relv + (size_t)(base + w) * DHEAD_ + d0;
        for (int i = 0; i < 32; ++i) rv_s[w * 64 + d0 + i] = (half_t)s[i];
    }
    wait_async();
    __builtin_amdgcn_s_wait_tensorcnt(0);
    __syncthreads();

    float*  mySs = sS + wv * 16 * 96;
    half_t* myP  = sP + wv * 16 * 104;
    float*  st_m = stats + wv * 64;
    float*  st_l = st_m + 16;
    float*  st_a = st_m + 32;
    const int mbase = wv * 16;
    const int ccol = lane & 15, crow = (lane >> 4) * 8;

    const v16h aq0 = frag_a(q_s + mbase * 72,      72, lane);
    const v16h aq1 = frag_a(q_s + mbase * 72 + 32, 72, lane);

    // ---------------- text branch: sim = Q K^T ----------------
    for (int t = 0; t < 5; ++t) {
        v8f acc = zero8f();
        acc = wmma32(aq0, frag_b(kt_s + t * 16 * 72,      72, lane), acc);
        acc = wmma32(aq1, frag_b(kt_s + t * 16 * 72 + 32, 72, lane), acc);
        for (int r = 0; r < 8; ++r) mySs[(crow + r) * 96 + t * 16 + ccol] = acc[r];
    }
    __syncthreads();

    // bias (Toeplitz rel_k) + softmax, one lane per query row
    if (lane < 16) {
        const int m  = lane;
        const int ml = mbase + m;
        const half_t* qrow = q_s + ml * 72;
        float mx = -3.4e38f;
        for (int j = 0; j < TEXTL_; ++j) {
            const half_t* rk = rk_s + (j - ml + 127) * 64;
            float bias = 0.f;
            for (int d = 0; d < 64; d += 8) {
                v8h qv = *(const v8h*)(qrow + d);
                v8h rv = *(const v8h*)(rk + d);
                for (int u = 0; u < 8; ++u) bias += (float)qv[u] * (float)rv[u];
            }
            float s = (mySs[m * 96 + j] + bias) * scale;
            mySs[m * 96 + j] = s;
            mx = fmaxf(mx, s);
        }
        float sum = 0.f;
        for (int j = 0; j < TEXTL_; ++j) {
            float p = __expf(mySs[m * 96 + j] - mx);
            mySs[m * 96 + j] = p; sum += p;
        }
        const float inv = 1.f / sum;
        for (int j = 0; j < TEXTL_; ++j) myP[m * 104 + j] = (half_t)(mySs[m * 96 + j] * inv);
        for (int j = TEXTL_; j < 96; ++j) myP[m * 104 + j] = (half_t)0;
    }
    __syncthreads();

    // O_text = P x V (P padded to K=96)
    v8f accO[4];
    for (int ni = 0; ni < 4; ++ni) accO[ni] = zero8f();
    for (int kc = 0; kc < 3; ++kc) {
        v16h pa = frag_a(myP + kc * 32, 104, lane);
        for (int ni = 0; ni < 4; ++ni)
            accO[ni] = wmma32(pa, frag_b(vt_s + ni * 16 * 88 + kc * 32, 88, lane), accO[ni]);
    }
    // + attn @ rel_v (Toeplitz, VALU from LDS)
    for (int r = 0; r < 8; ++r) {
        const int m = crow + r, ml = mbase + m;
        float a0 = 0, a1 = 0, a2 = 0, a3 = 0;
        for (int j = 0; j < TEXTL_; ++j) {
            const float p = (float)myP[m * 104 + j];
            const half_t* rv = rv_s + (j - ml + 127) * 64 + ccol;
            a0 += p * (float)rv[0];  a1 += p * (float)rv[16];
            a2 += p * (float)rv[32]; a3 += p * (float)rv[48];
        }
        accO[0][r] += a0; accO[1][r] += a1; accO[2][r] += a2; accO[3][r] += a3;
    }

    // ---------------- img branch: flash over 4 chunks of 64 keys ----------------
    if (lane < 16) { st_m[lane] = -3.4e38f; st_l[lane] = 0.f; }
    v8f accI[4];
    for (int ni = 0; ni < 4; ++ni) accI[ni] = zero8f();
    __syncthreads();

    for (int c = 0; c < 4; ++c) {
        for (int t = 0; t < 4; ++t) {
            const int j0 = c * 64 + t * 16;
            v8f acc = zero8f();
            acc = wmma32(aq0, frag_b(kip_s + j0 * 72,      72, lane), acc);
            acc = wmma32(aq1, frag_b(kip_s + j0 * 72 + 32, 72, lane), acc);
            for (int r = 0; r < 8; ++r)
                mySs[(crow + r) * 96 + t * 16 + ccol] = acc[r] * scale;
        }
        __syncthreads();
        if (lane < 16) {
            const int m = lane;
            const float mo = st_m[m];
            float cm = mo;
            for (int j = 0; j < 64; ++j) cm = fmaxf(cm, mySs[m * 96 + j]);
            const float al = __expf(mo - cm);
            float sum = 0.f;
            for (int j = 0; j < 64; ++j) {
                float p = __expf(mySs[m * 96 + j] - cm);
                myP[m * 104 + j] = (half_t)p; sum += p;
            }
            st_l[m] = st_l[m] * al + sum;
            st_m[m] = cm;
            st_a[m] = al;
        }
        __syncthreads();
        for (int ni = 0; ni < 4; ++ni)
            for (int r = 0; r < 8; ++r) accI[ni][r] *= st_a[crow + r];
        for (int kc = 0; kc < 2; ++kc) {
            v16h pa = frag_a(myP + kc * 32, 104, lane);
            for (int ni = 0; ni < 4; ++ni)
                accI[ni] = wmma32(pa,
                    frag_b(vip_s + ni * 16 * 264 + c * 64 + kc * 32, 264, lane), accI[ni]);
        }
        __syncthreads();
    }

    // ---------------- combine + store (merged-head f16) ----------------
    const float gate = tanhf(alpha_p[0]) + 1.0f;   // IMG_SCALE = 1.0
    for (int ni = 0; ni < 4; ++ni) {
        const int d = ni * 16 + ccol;
        for (int r = 0; r < 8; ++r) {
            const int m = crow + r;
            const float o = accO[ni][r] + gate * (accI[ni][r] / st_l[m]);
            const int gq = q0 + mbase + m;
            attn_out[((size_t)b * NQ_ + gq) * DMODEL_ + h * DHEAD_ + d] = (half_t)o;
        }
    }
}

// ---------------------------------------------------------------------------
// Output GEMM: d_out = attn(f16) x Wout^T(f16) + b_out  (f32 output)
// Both tiles are straight f16 copies -> async DMA to LDS.
// ---------------------------------------------------------------------------
__global__ __launch_bounds__(256)
void ca_out_gemm(const half_t* __restrict__ A, const half_t* __restrict__ Wt,
                 const float* __restrict__ bias, float* __restrict__ out) {
    __shared__ half_t As[128 * 40];
    __shared__ half_t Bs[128 * 40];

    const int tid = threadIdx.x, lane = tid & 31, wv = tid >> 5;
    const int m0 = blockIdx.x * 128, n0 = blockIdx.y * 128;
    const int wm = (wv & 3) * 32, wn = (wv >> 2) * 64;
    const int frow = tid >> 1, fc = (tid & 1) * 16;

    const half_t* arow = A  + (size_t)(m0 + frow) * DMODEL_;
    const half_t* brow = Wt + (size_t)(n0 + frow) * DMODEL_;
    const uint32_t as_lds = (uint32_t)(uintptr_t)(As + frow * 40 + fc);
    const uint32_t bs_lds = (uint32_t)(uintptr_t)(Bs + frow * 40 + fc);

    v8f acc[2][4];
    for (int mi = 0; mi < 2; ++mi)
        for (int ni = 0; ni < 4; ++ni) acc[mi][ni] = zero8f();

    for (int k0 = 0; k0 < DMODEL_; k0 += 32) {
        async_load_b128(as_lds,      arow + k0 + fc);
        async_load_b128(as_lds + 16, arow + k0 + fc + 8);
        async_load_b128(bs_lds,      brow + k0 + fc);
        async_load_b128(bs_lds + 16, brow + k0 + fc + 8);
        if (k0 + 128 < DMODEL_) __builtin_prefetch(arow + k0 + 128, 0, 0);
        wait_async();
        __syncthreads();

        v16h a0 = frag_a(As + wm * 40,        40, lane);
        v16h a1 = frag_a(As + (wm + 16) * 40, 40, lane);
        for (int ni = 0; ni < 4; ++ni) {
            v16h bf = frag_b(Bs + (wn + ni * 16) * 40, 40, lane);
            acc[0][ni] = wmma32(a0, bf, acc[0][ni]);
            acc[1][ni] = wmma32(a1, bf, acc[1][ni]);
        }
        __syncthreads();
    }

    const int ccol = lane & 15, crow = (lane >> 4) * 8;
    for (int mi = 0; mi < 2; ++mi)
        for (int ni = 0; ni < 4; ++ni)
            for (int r = 0; r < 8; ++r) {
                const int row = m0 + wm + mi * 16 + crow + r;
                const int col = n0 + wn + ni * 16 + ccol;
                out[(size_t)row * DMODEL_ + col] = acc[mi][ni][r] + bias[col];
            }
}

// ---------------------------------------------------------------------------
// Host launcher
// ---------------------------------------------------------------------------
extern "C" void kernel_launch(void* const* d_in, const int* in_sizes, int n_in,
                              void* d_out, int out_size, void* d_ws, size_t ws_size,
                              hipStream_t stream) {
    const float* x      = (const float*)d_in[0];
    const float* ctx    = (const float*)d_in[1];
    const float* Wq     = (const float*)d_in[2];
    const float* Wk     = (const float*)d_in[3];
    const float* Wv     = (const float*)d_in[4];
    const float* Wkip   = (const float*)d_in[5];
    const float* Wvip   = (const float*)d_in[6];
    const float* relk   = (const float*)d_in[7];
    const float* relv   = (const float*)d_in[8];
    const float* Wout   = (const float*)d_in[9];
    const float* bout   = (const float*)d_in[10];
    const float* alpha  = (const float*)d_in[11];
    (void)in_sizes; (void)n_in; (void)out_size; (void)ws_size;

    char* ws = (char*)d_ws;
    size_t off = 0;
    auto carve = [&](size_t bytes) -> void* {
        void* p = ws + off;
        off += (bytes + 255) & ~(size_t)255;
        return p;
    };
    const size_t WBYTES = (size_t)DMODEL_ * DMODEL_ * sizeof(half_t);
    half_t* Wq_t   = (half_t*)carve(WBYTES);
    half_t* Wk_t   = (half_t*)carve(WBYTES);
    half_t* Wv_t   = (half_t*)carve(WBYTES);
    half_t* Wkip_t = (half_t*)carve(WBYTES);
    half_t* Wvip_t = (half_t*)carve(WBYTES);
    half_t* Wout_t = (half_t*)carve(WBYTES);
    half_t* q_ws   = (half_t*)carve((size_t)NB_ * HEADS_ * NQ_   * DHEAD_ * sizeof(half_t));
    half_t* k_ws   = (half_t*)carve((size_t)NB_ * HEADS_ * TEXTL_ * DHEAD_ * sizeof(half_t));
    half_t* v_ws   = (half_t*)carve((size_t)NB_ * HEADS_ * TEXTL_ * DHEAD_ * sizeof(half_t));
    half_t* kip_ws = (half_t*)carve((size_t)NB_ * HEADS_ * IMGL_  * DHEAD_ * sizeof(half_t));
    half_t* vip_ws = (half_t*)carve((size_t)NB_ * HEADS_ * IMGL_  * DHEAD_ * sizeof(half_t));
    half_t* attn_ws= (half_t*)carve((size_t)NB_ * NQ_ * DMODEL_ * sizeof(half_t));

    // 1) weight convert + transpose
    ca_convert_w<<<4096, 256, 0, stream>>>(Wq,   Wq_t);
    ca_convert_w<<<4096, 256, 0, stream>>>(Wk,   Wk_t);
    ca_convert_w<<<4096, 256, 0, stream>>>(Wv,   Wv_t);
    ca_convert_w<<<4096, 256, 0, stream>>>(Wkip, Wkip_t);
    ca_convert_w<<<4096, 256, 0, stream>>>(Wvip, Wvip_t);
    ca_convert_w<<<4096, 256, 0, stream>>>(Wout, Wout_t);

    // 2) projections (head-split f16)
    ca_proj_gemm<<<dim3(256, 8), 256, 0, stream>>>(x,   Wq_t,   q_ws,   NQ_,   NQ_, 0,      NB_ * NQ_);
    ca_proj_gemm<<<dim3(20,  8), 256, 0, stream>>>(ctx, Wk_t,   k_ws,   TEXTL_, 333, 0,     NB_ * TEXTL_);
    ca_proj_gemm<<<dim3(20,  8), 256, 0, stream>>>(ctx, Wv_t,   v_ws,   TEXTL_, 333, 0,     NB_ * TEXTL_);
    ca_proj_gemm<<<dim3(64,  8), 256, 0, stream>>>(ctx, Wkip_t, kip_ws, IMGL_,  333, TEXTL_, NB_ * IMGL_);
    ca_proj_gemm<<<dim3(64,  8), 256, 0, stream>>>(ctx, Wvip_t, vip_ws, IMGL_,  333, TEXTL_, NB_ * IMGL_);

    // 3) fused attention (dynamic LDS: 241,920 B on the 320 KB WGP)
    const size_t smem = 241920;
    hipFuncSetAttribute(reinterpret_cast<const void*>(ca_attn),
                        hipFuncAttributeMaxDynamicSharedMemorySize, (int)smem);
    ca_attn<<<dim3(8, HEADS_, NB_), 256, smem, stream>>>(
        q_ws, k_ws, v_ws, kip_ws, vip_ws, relk, relv, alpha, attn_ws);

    // 4) output projection + bias (f32 out)
    ca_out_gemm<<<dim3(256, 8), 256, 0, stream>>>(attn_ws, Wout_t, bout, (float*)d_out);
}
